// AttnBlock_25881472926618
// MI455X (gfx1250) — matrix operand order repaired and checked by reading, not compiled, not run on hardware
//
#include <hip/hip_runtime.h>
#include <hip/hip_bf16.h>

typedef _Float16 h16;
typedef __attribute__((ext_vector_type(8)))  _Float16 v8h;
typedef __attribute__((ext_vector_type(16))) _Float16 v16h;
typedef __attribute__((ext_vector_type(8)))  float    v8f;

#define Ctot 512
#define Sdim 8192           // tokens per batch (8*32*32)
#define Ttot 16384          // B * Sdim

union U16h { v16h v; v8h h[2]; };

// A-operand tile (16x32 f16): lane = M%16 + 16*g ; K chunks at kofs+g*8 and kofs+16+g*8
__device__ inline v16h ld_a(const h16* rowptr, int g, int kofs) {
  U16h u;
  u.h[0] = *(const v8h*)(rowptr + kofs + g * 8);
  u.h[1] = *(const v8h*)(rowptr + kofs + 16 + g * 8);
  return u.v;
}
// B-operand tile (32x16 f16): lane = N%16 + 16*g ; K chunks at kofs+g*8 and kofs+16+g*8
__device__ inline v16h ld_b(const h16* colptr, int g, int kofs) {
  U16h u;
  u.h[0] = *(const v8h*)(colptr + kofs + g * 8);
  u.h[1] = *(const v8h*)(colptr + kofs + 16 + g * 8);
  return u.v;
}
__device__ inline v8f v8f_zero() {
  v8f z = {0.f, 0.f, 0.f, 0.f, 0.f, 0.f, 0.f, 0.f};
  return z;
}
__device__ inline v8f wmma_f16(v16h a, v16h b, v8f c) {
  return __builtin_amdgcn_wmma_f32_16x16x32_f16(false, a, false, b, (short)0, c,
                                                false, false);
}

// CDNA5 async copy: global -> LDS, 16 bytes per lane, tracked by ASYNCcnt.
// VDST VGPR carries the per-lane LDS byte offset.
__device__ inline void async_ld_b128(h16* lds_dst, const h16* gsrc) {
  int lofs = (int)(size_t)lds_dst;  // low 32 bits of generic ptr = LDS offset
  unsigned long long ga = (unsigned long long)(size_t)gsrc;
  asm volatile("global_load_async_to_lds_b128 %0, %1, off"
               :: "v"(lofs), "v"(ga) : "memory");
}
__device__ inline void async_wait0() {
  asm volatile("s_wait_asynccnt 0" ::: "memory");
}

// ---------------- f32 -> f16 weight conversion ----------------
__global__ void k_cvt(const float* __restrict__ a, h16* __restrict__ o, int n) {
  int i = blockIdx.x * 256 + threadIdx.x;
  if (i < n) o[i] = (h16)a[i];
}

// ---------------- RMS norm (channel dim) + transpose to token-major f16 ------
// block = 256 threads handles 16 tokens; x is [b][c][s] (c-stride = Sdim)
__global__ void k_norm(const float* __restrict__ x, const float* __restrict__ gamma,
                       h16* __restrict__ hn) {
  __shared__ float tile[16 * 513];
  __shared__ float red[16 * 16];
  __shared__ float inv[16];
  int t = threadIdx.x;
  int b = blockIdx.x >> 9;              // 512 blocks per batch
  int s0 = (blockIdx.x & 511) * 16;
  const float* xb = x + (long)b * Ctot * Sdim;
  // load 16 tokens x 512 ch into LDS (coalesced along s)
  for (int i = 0; i < 32; ++i) {
    int e = t + 256 * i;
    int s = e & 15, c = e >> 4;
    tile[s * 513 + c] = xb[(long)c * Sdim + s0 + s];
  }
  __syncthreads();
  // sum of squares: 16 threads per token
  {
    int s = t & 15, r = t >> 4;
    float acc = 0.f;
    for (int c = r * 32; c < r * 32 + 32; ++c) {
      float v = tile[s * 513 + c];
      acc += v * v;
    }
    red[s * 16 + r] = acc;
  }
  __syncthreads();
  if (t < 16) {
    float sum = 0.f;
    for (int r = 0; r < 16; ++r) sum += red[t * 16 + r];
    float n = fmaxf(sqrtf(sum), 1e-12f);
    inv[t] = 22.627416997969522f / n;   // sqrt(512)/||x||
  }
  __syncthreads();
  // write hn token-major (coalesced along c)
  for (int i = 0; i < 32; ++i) {
    int e = t + 256 * i;
    int c = e & 511, s = e >> 9;
    float v = tile[s * 513 + c] * inv[s] * gamma[c];
    hn[((long)b * Sdim + s0 + s) * Ctot + c] = (h16)v;
  }
}

// ---------------- QKV projection GEMM ----------------
// grid (8 Nstrips, 128), block 256 (8 waves); wave -> 16 tokens x 64 cols
// dst layout: transposed==0 -> [t][c] f16 ; transposed==1 -> [b][c][s] f16 (for V)
__global__ void k_qkv(const h16* __restrict__ hn, const h16* __restrict__ w,
                      const float* __restrict__ bias, h16* __restrict__ dst,
                      float scale, int transposed) {
  int lane = threadIdx.x & 31, wv = threadIdx.x >> 5;
  int g = lane >> 4, ln = lane & 15;
  long row0 = ((long)blockIdx.y * 8 + wv) * 16;
  int nbase = blockIdx.x * 64;
  v8f acc[4] = {v8f_zero(), v8f_zero(), v8f_zero(), v8f_zero()};
  const h16* arow = hn + (row0 + ln) * Ctot;
  for (int kt = 0; kt < 16; ++kt) {
    v16h a = ld_a(arow, g, kt * 32);
    for (int nt = 0; nt < 4; ++nt) {
      v16h bb = ld_b(w + (long)(nbase + nt * 16 + ln) * Ctot, g, kt * 32);
      acc[nt] = wmma_f16(a, bb, acc[nt]);
    }
  }
  for (int nt = 0; nt < 4; ++nt) {
    int n = nbase + nt * 16 + ln;
    float bia = bias[n];
    for (int r = 0; r < 8; ++r) {
      long tok = row0 + r + 8 * g;
      float v = (acc[nt][r] + bia) * scale;
      if (!transposed) {
        dst[tok * Ctot + n] = (h16)v;
      } else {
        long bb2 = tok >> 13, ss = tok & (Sdim - 1);
        dst[bb2 * ((long)Ctot * Sdim) + (long)n * Sdim + ss] = (h16)v;
      }
    }
  }
}

// ---------------- flash attention, block-causal over frames of 1024 ----------
// grid (512 qtiles, 2 batches), block 256 (8 waves). 16 queries per block.
// q,k: [t][c] f16 (q pre-scaled by 1/sqrt(C)) ; vT: [b][c][s] f16 ; ho: [t][c] f16
__global__ void k_attn(const h16* __restrict__ q, const h16* __restrict__ k,
                       const h16* __restrict__ vT, h16* __restrict__ ho) {
  __shared__ h16 q_lds[16 * 520];
  __shared__ h16 p_lds[16 * 136];
  __shared__ float pm[8 * 16], ps[8 * 16];
  __shared__ float mnew_s[16], arow_s[16], invl_s[16];

  int t = threadIdx.x, lane = t & 31, wv = t >> 5;
  int g = lane >> 4, ln = lane & 15;
  int b = blockIdx.y;
  long q0 = (long)blockIdx.x * 16;
  const h16* qb = q + ((long)b * Sdim + q0) * Ctot;
  // stage Q tile (16 x 512) into LDS via CDNA5 async-to-LDS b128 copies
  for (int i = 0; i < 4; ++i) {
    int ci = t + 256 * i;
    int row = ci >> 6, ch = (ci & 63) * 8;
    async_ld_b128(q_lds + row * 520 + ch, qb + (long)row * Ctot + ch);
  }
  async_wait0();
  __syncthreads();

  int qf = (int)(q0 >> 10);
  int kmax = (qf + 1) << 10;

  float m_old = -1e30f, l_old = 0.f, alpha_reg = 0.f;  // wave0 lanes 0..15: row state
  v8f oacc[4] = {v8f_zero(), v8f_zero(), v8f_zero(), v8f_zero()};
  const h16* kb_base = k + (long)b * Sdim * Ctot;
  const h16* vb_base = vT + (long)b * Ctot * Sdim;

  for (int kb = 0; kb < kmax; kb += 128) {
    // --- scores: this wave's 16 keys, full 512-channel dot product ---
    v8f sacc = v8f_zero();
    const h16* krow = kb_base + (long)(kb + wv * 16 + ln) * Ctot;
    __builtin_prefetch((const void*)(krow + 128 * Ctot), 0, 1);  // next key block
    for (int kt = 0; kt < 16; ++kt) {
      v16h a = ld_a(q_lds + ln * 520, g, kt * 32);
      v16h bb = ld_b(krow, g, kt * 32);
      sacc = wmma_f16(a, bb, sacc);
    }
    // per-wave row max (reduce over 16 lanes of each half)
    for (int r = 0; r < 8; ++r) {
      float v = sacc[r];
      v = fmaxf(v, __shfl_xor(v, 1, 32));
      v = fmaxf(v, __shfl_xor(v, 2, 32));
      v = fmaxf(v, __shfl_xor(v, 4, 32));
      v = fmaxf(v, __shfl_xor(v, 8, 32));
      if (ln == 0) pm[wv * 16 + r + 8 * g] = v;
    }
    __syncthreads();
    // --- wave0: online-softmax stats ---
    if (wv == 0 && lane < 16) {
      float mb = pm[lane];
      for (int w2 = 1; w2 < 8; ++w2) mb = fmaxf(mb, pm[w2 * 16 + lane]);
      float mn = fmaxf(m_old, mb);
      alpha_reg = __expf(m_old - mn);
      m_old = mn;
      mnew_s[lane] = mn;
      arow_s[lane] = alpha_reg;
    }
    __syncthreads();
    // --- all waves: P = exp(S - m), row sums, rescale O, stage P to LDS ---
    float mr[8], ar[8];
    for (int r = 0; r < 8; ++r) {
      mr[r] = mnew_s[r + 8 * g];
      ar[r] = arow_s[r + 8 * g];
    }
    for (int r = 0; r < 8; ++r) {
      float p = __expf(sacc[r] - mr[r]);
      p_lds[(r + 8 * g) * 136 + wv * 16 + ln] = (h16)p;
      float sum = p;
      sum += __shfl_xor(sum, 1, 32);
      sum += __shfl_xor(sum, 2, 32);
      sum += __shfl_xor(sum, 4, 32);
      sum += __shfl_xor(sum, 8, 32);
      if (ln == 0) ps[wv * 16 + r + 8 * g] = sum;
    }
    for (int nt = 0; nt < 4; ++nt)
      for (int r = 0; r < 8; ++r) oacc[nt][r] *= ar[r];
    __syncthreads();
    // --- wave0: l update ; all waves: O += P @ V (64-ch slice per wave) ---
    if (wv == 0 && lane < 16) {
      float s2 = 0.f;
      for (int w2 = 0; w2 < 8; ++w2) s2 += ps[w2 * 16 + lane];
      l_old = l_old * alpha_reg + s2;
    }
    for (int kt = 0; kt < 4; ++kt) {
      v16h a = ld_a(p_lds + ln * 136, g, kt * 32);
      for (int nt = 0; nt < 4; ++nt) {
        const h16* vcol = vb_base + (long)(wv * 64 + nt * 16 + ln) * Sdim + kb;
        v16h bb = ld_b(vcol, g, kt * 32);
        oacc[nt] = wmma_f16(a, bb, oacc[nt]);
      }
    }
    __syncthreads();
  }
  if (wv == 0 && lane < 16) invl_s[lane] = 1.0f / l_old;
  __syncthreads();
  for (int nt = 0; nt < 4; ++nt) {
    int ch = wv * 64 + nt * 16 + ln;
    for (int r = 0; r < 8; ++r) {
      int row = r + 8 * g;
      float v = oacc[nt][r] * invl_s[row];
      ho[((long)b * Sdim + q0 + row) * Ctot + ch] = (h16)v;
    }
  }
}

// ---------------- output projection + bias + residual ----------------
// out[b][c][s] f32 = x + ho @ wo^T + bo ; grid (8,128), block 256
__global__ void k_out(const h16* __restrict__ ho, const h16* __restrict__ wo,
                      const float* __restrict__ bo, const float* __restrict__ x,
                      float* __restrict__ out) {
  int lane = threadIdx.x & 31, wv = threadIdx.x >> 5;
  int g = lane >> 4, ln = lane & 15;
  long row0 = ((long)blockIdx.y * 8 + wv) * 16;
  int nbase = blockIdx.x * 64;
  v8f acc[4] = {v8f_zero(), v8f_zero(), v8f_zero(), v8f_zero()};
  const h16* arow = ho + (row0 + ln) * Ctot;
  for (int kt = 0; kt < 16; ++kt) {
    v16h a = ld_a(arow, g, kt * 32);
    for (int nt = 0; nt < 4; ++nt) {
      v16h bb = ld_b(wo + (long)(nbase + nt * 16 + ln) * Ctot, g, kt * 32);
      acc[nt] = wmma_f16(a, bb, acc[nt]);
    }
  }
  for (int nt = 0; nt < 4; ++nt) {
    int n = nbase + nt * 16 + ln;
    float bia = bo[n];
    for (int r = 0; r < 8; ++r) {
      long tok = row0 + r + 8 * g;
      long bb2 = tok >> 13, ss = tok & (Sdim - 1);
      long oidx = bb2 * ((long)Ctot * Sdim) + (long)n * Sdim + ss;
      out[oidx] = x[oidx] + acc[nt][r] + bia;
    }
  }
}

extern "C" void kernel_launch(void* const* d_in, const int* in_sizes, int n_in,
                              void* d_out, int out_size, void* d_ws, size_t ws_size,
                              hipStream_t stream) {
  const float* x     = (const float*)d_in[0];
  const float* gamma = (const float*)d_in[1];
  const float* wq    = (const float*)d_in[2];
  const float* bq    = (const float*)d_in[3];
  const float* wk    = (const float*)d_in[4];
  const float* bk    = (const float*)d_in[5];
  const float* wv    = (const float*)d_in[6];
  const float* bv    = (const float*)d_in[7];
  const float* wo    = (const float*)d_in[8];
  const float* bo    = (const float*)d_in[9];
  float* out = (float*)d_out;

  const long SC = (long)Ttot * Ctot;   // 16384*512 halves per tensor
  h16* ws   = (h16*)d_ws;
  h16* hn   = ws;                       // [t][c]   (reused as ho after QKV)
  h16* qf16 = hn + SC;                  // [t][c], pre-scaled by 1/sqrt(C)
  h16* kf16 = qf16 + SC;                // [t][c]
  h16* vT16 = kf16 + SC;                // [b][c][s]
  h16* w16  = vT16 + SC;                // 4 x 512*512 f16 weights
  h16* wq16 = w16;
  h16* wk16 = w16 + 262144;
  h16* wv16 = w16 + 2 * 262144;
  h16* wo16 = w16 + 3 * 262144;
  h16* ho16 = hn;                       // alias: hn dead after QKV GEMMs

  const int WN = 262144;
  k_cvt<<<WN / 256, 256, 0, stream>>>(wq, wq16, WN);
  k_cvt<<<WN / 256, 256, 0, stream>>>(wk, wk16, WN);
  k_cvt<<<WN / 256, 256, 0, stream>>>(wv, wv16, WN);
  k_cvt<<<WN / 256, 256, 0, stream>>>(wo, wo16, WN);

  k_norm<<<1024, 256, 0, stream>>>(x, gamma, hn);

  const float qscale = 0.044194173824159216f;  // 1/sqrt(512)
  k_qkv<<<dim3(8, 128), 256, 0, stream>>>(hn, wq16, bq, qf16, qscale, 0);
  k_qkv<<<dim3(8, 128), 256, 0, stream>>>(hn, wk16, bk, kf16, 1.0f, 0);
  k_qkv<<<dim3(8, 128), 256, 0, stream>>>(hn, wv16, bv, vT16, 1.0f, 1);

  k_attn<<<dim3(512, 2), 256, 0, stream>>>(qf16, kf16, vT16, ho16);

  k_out<<<dim3(8, 128), 256, 0, stream>>>(ho16, wo16, bo, x, out);
}
